// PointNet2_cls_37469294691167
// MI455X (gfx1250) — matrix operand order repaired and checked by reading, not compiled, not run on hardware
//
#include <hip/hip_runtime.h>

typedef __bf16 bf16x16 __attribute__((ext_vector_type(16)));
typedef float f32x8 __attribute__((ext_vector_type(8)));

union FragAB { bf16x16 v; uint4 u[2]; };

__device__ __forceinline__ unsigned short f2bf(float f) {
    unsigned int u = __float_as_uint(f);
    unsigned int r = (u + 0x7FFFu + ((u >> 16) & 1u)) >> 16;
    return (unsigned short)r;
}
__device__ __forceinline__ float bf2f(unsigned short h) {
    return __uint_as_float(((unsigned int)h) << 16);
}

// ---------------------------------------------------------------------------
// xyz (B,3,N) -> xyz_t (B,N,3)
__global__ __launch_bounds__(256) void k_transpose_xyz(const float* __restrict__ xyz,
                                                       float* __restrict__ xyz_t,
                                                       int Bn, int N) {
    int t = blockIdx.x * 256 + threadIdx.x;
    int total = Bn * N * 3;
    if (t >= total) return;
    int b = t / (N * 3);
    int r = t - b * N * 3;
    int n = r / 3;
    int c = r - n * 3;
    xyz_t[t] = xyz[(b * 3 + c) * N + n];
}

// ---------------------------------------------------------------------------
// Farthest point sampling: one block per batch, dist kept in LDS.
__global__ __launch_bounds__(256) void k_fps(const float* __restrict__ xyz_t,
                                             int* __restrict__ fpsIdx,
                                             int N, int npoint) {
    __shared__ float dist[2048];
    __shared__ float rv[256];
    __shared__ int   ri[256];
    __shared__ int   sFar;
    int b = blockIdx.x, tid = threadIdx.x;
    for (int j = tid; j < N; j += 256) dist[j] = 1e10f;
    if (tid == 0) sFar = 0;
    __syncthreads();
    for (int it = 0; it < npoint; ++it) {
        int far = sFar;
        if (tid == 0) fpsIdx[b * npoint + it] = far;
        float cx = xyz_t[(b * N + far) * 3 + 0];
        float cy = xyz_t[(b * N + far) * 3 + 1];
        float cz = xyz_t[(b * N + far) * 3 + 2];
        float bestv = -1.f; int besti = N;
        for (int j = tid; j < N; j += 256) {
            float dx = xyz_t[(b * N + j) * 3 + 0] - cx;
            float dy = xyz_t[(b * N + j) * 3 + 1] - cy;
            float dz = xyz_t[(b * N + j) * 3 + 2] - cz;
            float d = dx * dx + dy * dy + dz * dz;
            float nd = fminf(dist[j], d);
            dist[j] = nd;
            if (nd > bestv || (nd == bestv && j < besti)) { bestv = nd; besti = j; }
        }
        rv[tid] = bestv; ri[tid] = besti;
        __syncthreads();
        for (int s = 128; s > 0; s >>= 1) {
            if (tid < s) {
                if (rv[tid + s] > rv[tid] || (rv[tid + s] == rv[tid] && ri[tid + s] < ri[tid])) {
                    rv[tid] = rv[tid + s]; ri[tid] = ri[tid + s];
                }
            }
            __syncthreads();
        }
        if (tid == 0) sFar = ri[0];
        __syncthreads();
    }
}

// ---------------------------------------------------------------------------
// Gather centers: new_xyz (B,S,3) = xyz_t[fpsIdx]
__global__ __launch_bounds__(256) void k_gather_centers(const float* __restrict__ xyz_t,
                                                        const int* __restrict__ fpsIdx,
                                                        float* __restrict__ newXyz,
                                                        int G, int S, int N) {
    int t = blockIdx.x * 256 + threadIdx.x;
    if (t >= G * 3) return;
    int g = t / 3, c = t - g * 3;
    int b = g / S;
    int p = fpsIdx[g];
    newXyz[t] = xyz_t[(b * N + p) * 3 + c];
}

// ---------------------------------------------------------------------------
// Ball query: one wave32 per group; collects first K indices with d^2 <= r^2
// (in ascending index order, matching the reference's sort), pads with first.
__global__ __launch_bounds__(128) void k_ball_query(const float* __restrict__ xyz_t,
                                                    const float* __restrict__ centers,
                                                    int* __restrict__ idxOut,
                                                    int G, int S, int N, int K, float r2) {
    int wave = threadIdx.x >> 5;
    int lane = threadIdx.x & 31;
    int g = blockIdx.x * 4 + wave;
    if (g >= G) return;
    int b = g / S;
    float cx = centers[g * 3 + 0];
    float cy = centers[g * 3 + 1];
    float cz = centers[g * 3 + 2];
    int count = 0;
    int firstIdx = 0;
    bool haveFirst = false;
    for (int base = 0; base < N && count < K; base += 32) {
        int j = base + lane;
        bool in = false;
        if (j < N) {
            float dx = xyz_t[(b * N + j) * 3 + 0] - cx;
            float dy = xyz_t[(b * N + j) * 3 + 1] - cy;
            float dz = xyz_t[(b * N + j) * 3 + 2] - cz;
            in = (dx * dx + dy * dy + dz * dz) <= r2;
        }
        unsigned long long mall = __ballot(in);
        unsigned int mm = (unsigned int)mall;
        if (!haveFirst && mm) { firstIdx = base + __builtin_ctz(mm); haveFirst = true; }
        int prefix = __popc(mm & ((lane == 0) ? 0u : ((1u << lane) - 1u)));
        int slot = count + prefix;
        if (in && slot < K) idxOut[g * K + slot] = j;
        count += __popc(mm);
    }
    if (count < K) {
        for (int p = count + lane; p < K; p += 32) idxOut[g * K + p] = firstIdx;
    }
}

// ---------------------------------------------------------------------------
// Pack W(cin,cout) f32 -> Wt(coutP x cinP) bf16, zero padded (B-fragment friendly)
__global__ __launch_bounds__(256) void k_pack_wt(const float* __restrict__ W,
                                                 unsigned short* __restrict__ Wt,
                                                 int cin, int cout, int cinP, int coutP) {
    int t = blockIdx.x * 256 + threadIdx.x;
    if (t >= cinP * coutP) return;
    int o = t / cinP, i = t - o * cinP;
    float v = (o < cout && i < cin) ? W[i * cout + o] : 0.f;
    Wt[t] = f2bf(v);
}

// ---------------------------------------------------------------------------
// Fused grouped MLP (3 layers) + maxpool, all activations resident in LDS.
// One 256-thread block (8 waves) per (group, sub) pair.
// Matmuls: v_wmma_f32_16x16x32_bf16 with 2x2 (32x32) register blocking per
// wave: 2 A-frags + 2 B-frags -> 4 WMMAs per K-step (halves fragment traffic,
// 4 independent accumulation chains hide XDL latency).
__global__ __launch_bounds__(256) void k_sa_group_mlp(
    const float* __restrict__ xyz_t,   // (B, Npts, 3)
    const float* __restrict__ pts,     // (B, Npts, Cpts)
    const int* __restrict__ idx,       // (B*S, Kfull) or nullptr -> identity
    const float* __restrict__ newXyz,  // (B*S, 3) or nullptr -> no centering
    const unsigned short* __restrict__ w0, const float* __restrict__ b0,
    const unsigned short* __restrict__ w1, const float* __restrict__ b1,
    const unsigned short* __restrict__ w2, const float* __restrict__ b2,
    float* __restrict__ outPartial,    // (B*S*nsub, c3)
    int S, int Npts, int Cpts, int xyzFirst,
    int Kfull, int nsub, int Ksub,
    int cinP, int c1, int c1P, int c2, int c2P, int c3, int c3P)
{
    extern __shared__ unsigned short smem[];
    int maxW = cinP;
    if (c1P > maxW) maxW = c1P;
    if (c2P > maxW) maxW = c2P;
    if (c3P > maxW) maxW = c3P;
    unsigned short* bufA = smem;
    unsigned short* bufB = smem + Ksub * maxW;

    int tid = threadIdx.x;
    int lane = tid & 31;
    int waveId = tid >> 5;

    int gid = blockIdx.x;
    int sub = gid % nsub;
    int gs = gid / nsub;
    int s = gs % S;
    int b = gs / S;

    // ---- stage input tile: (Ksub x cinP) bf16, zero-padded channels ----
    for (int t = tid; t < Ksub * cinP; t += 256) {
        int k = t / cinP, c = t - k * cinP;
        int kk = sub * Ksub + k;
        int p = idx ? idx[gs * Kfull + kk] : (s * Kfull + kk);
        float val = 0.f;
        if (xyzFirst) {
            if (c < 3) {
                val = xyz_t[(b * Npts + p) * 3 + c];
                if (newXyz) val -= newXyz[gs * 3 + c];
            } else if (c - 3 < Cpts) {
                val = pts[(b * Npts + p) * Cpts + (c - 3)];
            }
        } else {
            if (c < Cpts) {
                val = pts[(b * Npts + p) * Cpts + c];
            } else if (c - Cpts < 3) {
                int cc = c - Cpts;
                val = xyz_t[(b * Npts + p) * 3 + cc];
                if (newXyz) val -= newXyz[gs * 3 + cc];
            }
        }
        bufA[k * cinP + c] = f2bf(val);
    }
    __syncthreads();

    const unsigned short* Ws[3] = { w0, w1, w2 };
    const float* Bs[3] = { b0, b1, b2 };
    int inPs[3]  = { cinP, c1P, c2P };
    int outRs[3] = { c1, c2, c3 };
    int outPs[3] = { c1P, c2P, c3P };

#pragma unroll
    for (int L = 0; L < 3; ++L) {
        const unsigned short* src = (L & 1) ? bufB : bufA;
        unsigned short* dst = (L & 1) ? bufA : bufB;
        int inP = inPs[L], oR = outRs[L], oP = outPs[L];
        const unsigned short* W = Ws[L];
        const float* bias = Bs[L];
        // 2x2 blocking: Mt, Nt are always even (Ksub mult of 32 rows? Ksub>=32,
        // widths padded to mult of 32) -> units = (Mt/2)*(Nt/2), each 32x32.
        int Mu = Ksub >> 5, Nu = oP >> 5;
        int units = Mu * Nu;
        for (int t = waveId; t < units; t += 8) {
            int um = t / Nu;
            int un = t - um * Nu;
            int m0 = um << 5;
            int n0 = un << 5;
            int kha = (lane >> 4) << 3;                 // A: 0 or 8
            int khb = (lane >> 4) << 3;                 // B: 0 or 8
            const unsigned short* aBase0 = src + (m0 + (lane & 15)) * inP + kha;
            const unsigned short* aBase1 = aBase0 + 16 * inP;
            const unsigned short* bBase0 = W + (n0 + (lane & 15)) * inP + khb;
            const unsigned short* bBase1 = bBase0 + 16 * inP;
            f32x8 acc00 = { 0.f, 0.f, 0.f, 0.f, 0.f, 0.f, 0.f, 0.f };
            f32x8 acc01 = acc00, acc10 = acc00, acc11 = acc00;
            for (int k0 = 0; k0 < inP; k0 += 32) {
                FragAB a0, a1, bb0, bb1;
                a0.u[0]  = *(const uint4*)(aBase0 + k0);
                a0.u[1]  = *(const uint4*)(aBase0 + k0 + 16);
                a1.u[0]  = *(const uint4*)(aBase1 + k0);
                a1.u[1]  = *(const uint4*)(aBase1 + k0 + 16);
                bb0.u[0] = *(const uint4*)(bBase0 + k0);
                bb0.u[1] = *(const uint4*)(bBase0 + k0 + 16);
                bb1.u[0] = *(const uint4*)(bBase1 + k0);
                bb1.u[1] = *(const uint4*)(bBase1 + k0 + 16);
                acc00 = __builtin_amdgcn_wmma_f32_16x16x32_bf16(
                    false, a0.v, false, bb0.v, (short)0, acc00, false, false);
                acc01 = __builtin_amdgcn_wmma_f32_16x16x32_bf16(
                    false, a0.v, false, bb1.v, (short)0, acc01, false, false);
                acc10 = __builtin_amdgcn_wmma_f32_16x16x32_bf16(
                    false, a1.v, false, bb0.v, (short)0, acc10, false, false);
                acc11 = __builtin_amdgcn_wmma_f32_16x16x32_bf16(
                    false, a1.v, false, bb1.v, (short)0, acc11, false, false);
            }
            // epilogue: C/D layout -> bias + relu -> LDS bf16
            int col0 = n0 + (lane & 15);
            int col1 = col0 + 16;
            float bv0 = (col0 < oR) ? bias[col0] : 0.f;
            float bv1 = (col1 < oR) ? bias[col1] : 0.f;
            int rb0 = m0 + ((lane >> 4) << 3);
            int rb1 = rb0 + 16;
#pragma unroll
            for (int r = 0; r < 8; ++r) {
                float v00 = fmaxf(acc00[r] + bv0, 0.f);
                float v01 = fmaxf(acc01[r] + bv1, 0.f);
                float v10 = fmaxf(acc10[r] + bv0, 0.f);
                float v11 = fmaxf(acc11[r] + bv1, 0.f);
                dst[(rb0 + r) * oP + col0] = f2bf(v00);
                dst[(rb0 + r) * oP + col1] = f2bf(v01);
                dst[(rb1 + r) * oP + col0] = f2bf(v10);
                dst[(rb1 + r) * oP + col1] = f2bf(v11);
            }
        }
        __syncthreads();
    }

    // ---- max-pool over Ksub rows (layer 3 output lives in bufB) ----
    for (int c = tid; c < c3; c += 256) {
        float m = -3.4e38f;
        for (int k = 0; k < Ksub; ++k) m = fmaxf(m, bf2f(bufB[k * c3P + c]));
        outPartial[(long)gid * c3 + c] = m;
    }
}

// ---------------------------------------------------------------------------
// Combine nsub partial maxima and scatter into concat destination.
__global__ __launch_bounds__(256) void k_combine_max(const float* __restrict__ part,
                                                     float* __restrict__ dst,
                                                     int G, int nsub, int C,
                                                     int dstStride, int chanOff) {
    int t = blockIdx.x * 256 + threadIdx.x;
    if (t >= G * C) return;
    int g = t / C, c = t - g * C;
    float m = part[(long)(g * nsub) * C + c];
    for (int s = 1; s < nsub; ++s) m = fmaxf(m, part[(long)(g * nsub + s) * C + c]);
    dst[(long)g * dstStride + chanOff + c] = m;
}

// ---------------------------------------------------------------------------
// Tiny dense head (fp32). mode 1 = leaky relu(0.01), 0 = identity.
__global__ __launch_bounds__(256) void k_dense(const float* __restrict__ x,
                                               const float* __restrict__ W,
                                               const float* __restrict__ bias,
                                               float* __restrict__ y,
                                               int Bn, int Cin, int Cout, int mode) {
    int t = blockIdx.x * 256 + threadIdx.x;
    if (t >= Bn * Cout) return;
    int b = t / Cout, o = t - b * Cout;
    float acc = bias[o];
    const float* xr = x + b * Cin;
    for (int i = 0; i < Cin; ++i) acc += xr[i] * W[i * Cout + o];
    if (mode == 1) acc = (acc > 0.f) ? acc : 0.01f * acc;
    y[t] = acc;
}

// ===========================================================================
extern "C" void kernel_launch(void* const* d_in, const int* in_sizes, int n_in,
                              void* d_out, int out_size, void* d_ws, size_t ws_size,
                              hipStream_t stream) {
    (void)in_sizes; (void)n_in; (void)out_size; (void)ws_size;
    const int B = 32, N = 2048, S1 = 512, S2 = 128;
    auto ceil32 = [](int x) { return (x + 31) & ~31; };

    char* base = (char*)d_ws;
    size_t off = 0;
    auto alloc = [&](size_t bytes) -> void* {
        void* p = base + off;
        off += (bytes + 255) & ~(size_t)255;
        return p;
    };

    float* xyzT = (float*)alloc((size_t)B * N * 3 * 4);
    int*   fps1 = (int*)alloc((size_t)B * S1 * 4);
    float* nx1  = (float*)alloc((size_t)B * S1 * 3 * 4);
    int*   fps2 = (int*)alloc((size_t)B * S2 * 4);
    float* nx2  = (float*)alloc((size_t)B * S2 * 3 * 4);
    int*   idxB = (int*)alloc((size_t)B * S1 * 128 * 4);   // reused per branch (stream-serialized)
    float* l1   = (float*)alloc((size_t)B * S1 * 320 * 4);
    float* l2   = (float*)alloc((size_t)B * S2 * 512 * 4);
    float* part = (float*)alloc((size_t)B * S1 * 128 * 4); // largest partial use
    float* feat = (float*)alloc((size_t)B * 1024 * 4);
    float* hid  = (float*)alloc((size_t)B * 512 * 4);

    // layer table: d_in index of W, cin, cout (bias is at widx+1)
    struct LayerT { int wi, ci, co; };
    const LayerT Ls[18] = {
        {1, 6, 32},   {3, 32, 32},   {5, 32, 64},     // sa1 mlp0
        {7, 6, 64},   {9, 64, 64},   {11, 64, 128},   // sa1 mlp1
        {13, 6, 64},  {15, 64, 96},  {17, 96, 128},   // sa1 mlp2
        {19, 323, 128}, {21, 128, 128}, {23, 128, 256}, // sa2 mlp0
        {25, 323, 128}, {27, 128, 196}, {29, 196, 256}, // sa2 mlp1
        {31, 515, 256}, {33, 256, 512}, {35, 512, 1024} // sa3
    };
    unsigned short* wt[18];
    for (int i = 0; i < 18; ++i) {
        int cip = ceil32(Ls[i].ci), cop = ceil32(Ls[i].co);
        wt[i] = (unsigned short*)alloc((size_t)cip * cop * 2);
    }

    (void)hipFuncSetAttribute(reinterpret_cast<const void*>(k_sa_group_mlp),
                              hipFuncAttributeMaxDynamicSharedMemorySize, 262144);

    // ---- pack weights to bf16 transposed/padded ----
    for (int i = 0; i < 18; ++i) {
        int cip = ceil32(Ls[i].ci), cop = ceil32(Ls[i].co);
        int total = cip * cop;
        k_pack_wt<<<(total + 255) / 256, 256, 0, stream>>>(
            (const float*)d_in[Ls[i].wi], wt[i], Ls[i].ci, Ls[i].co, cip, cop);
    }

    // ---- transpose xyz ----
    {
        int total = B * N * 3;
        k_transpose_xyz<<<(total + 255) / 256, 256, 0, stream>>>(
            (const float*)d_in[0], xyzT, B, N);
    }

    // ---- stage 1: FPS + centers ----
    k_fps<<<B, 256, 0, stream>>>(xyzT, fps1, N, S1);
    {
        int total = B * S1 * 3;
        k_gather_centers<<<(total + 255) / 256, 256, 0, stream>>>(xyzT, fps1, nx1, B * S1, S1, N);
    }

    // SA1 branches: (radius^2, K, layer base index, channel offset, out channels)
    struct BrT { float r2; int K; int l0; int chOff; int C; };
    const BrT sa1[3] = { {0.01f, 32, 0, 0, 64}, {0.04f, 64, 3, 64, 128}, {0.16f, 128, 6, 192, 128} };
    for (int br = 0; br < 3; ++br) {
        const BrT& R = sa1[br];
        int G = B * S1;
        k_ball_query<<<(G + 3) / 4, 128, 0, stream>>>(xyzT, nx1, idxB, G, S1, N, R.K, R.r2);
        int l0 = R.l0;
        int c1 = Ls[l0].co, c2 = Ls[l0 + 1].co, c3 = Ls[l0 + 2].co;
        int c1P = ceil32(c1), c2P = ceil32(c2), c3P = ceil32(c3);
        int cinP = 32; // cin = 6 (3 pts feats + 3 xyz), padded
        int maxW = cinP;
        if (c1P > maxW) maxW = c1P; if (c2P > maxW) maxW = c2P; if (c3P > maxW) maxW = c3P;
        size_t smem = (size_t)2 * R.K * maxW * 2;
        k_sa_group_mlp<<<G, 256, smem, stream>>>(
            xyzT, xyzT, idxB, nx1,
            wt[l0], (const float*)d_in[Ls[l0].wi + 1],
            wt[l0 + 1], (const float*)d_in[Ls[l0 + 1].wi + 1],
            wt[l0 + 2], (const float*)d_in[Ls[l0 + 2].wi + 1],
            part,
            S1, N, 3, /*xyzFirst=*/0,
            R.K, /*nsub=*/1, /*Ksub=*/R.K,
            cinP, c1, c1P, c2, c2P, c3, c3P);
        int total = G * R.C;
        k_combine_max<<<(total + 255) / 256, 256, 0, stream>>>(part, l1, G, 1, R.C, 320, R.chOff);
    }

    // ---- stage 2: FPS over l1 centers ----
    k_fps<<<B, 256, 0, stream>>>(nx1, fps2, S1, S2);
    {
        int total = B * S2 * 3;
        k_gather_centers<<<(total + 255) / 256, 256, 0, stream>>>(nx1, fps2, nx2, B * S2, S2, S1);
    }

    const BrT sa2[2] = { {0.16f, 64, 9, 0, 256}, {0.64f, 128, 12, 256, 256} };
    for (int br = 0; br < 2; ++br) {
        const BrT& R = sa2[br];
        int G = B * S2;
        k_ball_query<<<(G + 3) / 4, 128, 0, stream>>>(nx1, nx2, idxB, G, S2, S1, R.K, R.r2);
        int l0 = R.l0;
        int c1 = Ls[l0].co, c2 = Ls[l0 + 1].co, c3 = Ls[l0 + 2].co;
        int c1P = ceil32(c1), c2P = ceil32(c2), c3P = ceil32(c3);
        int cinP = ceil32(320 + 3); // 352
        int maxW = cinP;
        if (c1P > maxW) maxW = c1P; if (c2P > maxW) maxW = c2P; if (c3P > maxW) maxW = c3P;
        size_t smem = (size_t)2 * R.K * maxW * 2;
        k_sa_group_mlp<<<G, 256, smem, stream>>>(
            nx1, l1, idxB, nx2,
            wt[l0], (const float*)d_in[Ls[l0].wi + 1],
            wt[l0 + 1], (const float*)d_in[Ls[l0 + 1].wi + 1],
            wt[l0 + 2], (const float*)d_in[Ls[l0 + 2].wi + 1],
            part,
            S2, S1, 320, /*xyzFirst=*/0,
            R.K, /*nsub=*/1, /*Ksub=*/R.K,
            cinP, c1, c1P, c2, c2P, c3, c3P);
        int total = G * R.C;
        k_combine_max<<<(total + 255) / 256, 256, 0, stream>>>(part, l2, G, 1, R.C, 512, R.chOff);
    }

    // ---- stage 3: global MLP [515 -> 256 -> 512 -> 1024] + global max ----
    {
        int l0 = 15;
        int c1 = 256, c2 = 512, c3 = 1024;
        int cinP = ceil32(512 + 3); // 544
        size_t smem = (size_t)2 * 64 * 1024 * 2; // 256 KB (Ksub=64, maxW=1024)
        k_sa_group_mlp<<<B * 2, 256, smem, stream>>>(
            nx2, l2, nullptr, nullptr,
            wt[l0], (const float*)d_in[Ls[l0].wi + 1],
            wt[l0 + 1], (const float*)d_in[Ls[l0 + 1].wi + 1],
            wt[l0 + 2], (const float*)d_in[Ls[l0 + 2].wi + 1],
            part,
            /*S=*/1, /*Npts=*/S2, /*Cpts=*/512, /*xyzFirst=*/1,
            /*Kfull=*/S2, /*nsub=*/2, /*Ksub=*/64,
            cinP, c1, c1, c2, c2, c3, c3);
        int total = B * 1024;
        k_combine_max<<<(total + 255) / 256, 256, 0, stream>>>(part, feat, B, 2, 1024, 1024, 0);
    }

    // ---- head: 1024 -> 512 (leaky 0.01) -> 40 ----
    {
        int total = B * 512;
        k_dense<<<(total + 255) / 256, 256, 0, stream>>>(
            feat, (const float*)d_in[37], (const float*)d_in[38], hid, B, 1024, 512, 1);
        total = B * 40;
        k_dense<<<(total + 255) / 256, 256, 0, stream>>>(
            hid, (const float*)d_in[39], (const float*)d_in[40], (float*)d_out, B, 512, 40, 0);
    }
}
